// FeatureAttnNet_30906584662249
// MI455X (gfx1250) — matrix operand order repaired and checked by reading, not compiled, not run on hardware
//
#include <hip/hip_runtime.h>
#include <hip/hip_bf16.h>
#include <math.h>

// ---------------------------------------------------------------------------
// FeatureAttnNet for MI455X (gfx1250, wave32, WMMA + TDM).
//
// Roofline: ~43 GFLOP of matmul vs ~63 MB of HBM traffic -> compute-bound.
// Strategy: convert fp32 operands to bf16 once (weights pre-transposed to
// [col][k] so all LDS staging is vectorized), run all matmuls through
// v_wmma_f32_16x16x32_bf16 with fp32 accumulation; LN / softmax in fp32.
// GEMM: 128x64 block tile, double-buffered LDS + register prefetch; each
// staged B tile feeds 8 WMMAs/wave between barriers. Attention: Fi tile is
// DMA'd into LDS by the Tensor Data Mover (row padding done by TDM pad
// logic), synchronized with s_wait_tensorcnt.
// ---------------------------------------------------------------------------

typedef __attribute__((ext_vector_type(16))) __bf16 v16bf;
typedef __attribute__((ext_vector_type(8)))  float  v8f;
typedef __attribute__((ext_vector_type(4)))  unsigned int u32x4;
typedef __attribute__((ext_vector_type(8)))  int          i32x8;
typedef __attribute__((ext_vector_type(4)))  int          i32x4;

struct alignas(16) U4    { unsigned int a, b, c, d; };       // 16B copy unit
struct alignas(16) B16x8 { unsigned short u[8]; };           // 8 x bf16
union  Frag { B16x8 h[2]; v16bf v; unsigned short u[16]; };  // 16 x bf16 = 32B

// Problem constants (match reference)
#define BB     1024
#define N_ENC  2048
#define N_HID  1024
#define N_CMD  128
#define DD     64
#define HH     8
#define NFNF   128
#define MOUT   4096     // M = D*NF/2

__device__ __forceinline__ unsigned short f2bf(float f) {
    unsigned int u = __float_as_uint(f);
    unsigned int r = u + 0x7fffu + ((u >> 16) & 1u);   // round-to-nearest-even
    return (unsigned short)(r >> 16);
}
__device__ __forceinline__ float bf2f(unsigned short h) {
    return __uint_as_float(((unsigned int)h) << 16);
}
__device__ __forceinline__ float gelu_exact(float x) {
    return 0.5f * x * (1.0f + erff(x * 0.70710678118654752f));
}

// ---------------------------------------------------------------------------
// Kernel A1: fp32 -> bf16 elementwise (activations)
// ---------------------------------------------------------------------------
__global__ __launch_bounds__(256) void k_cvt(const float* __restrict__ s,
                                             unsigned short* __restrict__ d, int n) {
    int i = blockIdx.x * blockDim.x + threadIdx.x;
    if (i < n) d[i] = f2bf(s[i]);
}

// ---------------------------------------------------------------------------
// Kernel A2: big weights [K, 4096] fp32 -> bf16 transposed [4096, K].
// LDS-tiled 64x64 so both the read and the write are coalesced.
// ---------------------------------------------------------------------------
__global__ __launch_bounds__(256) void k_cvt_T(const float* __restrict__ src,   // [K][4096]
                                               unsigned short* __restrict__ dst, // [4096][K]
                                               int K)
{
    __shared__ float tile[64][65];
    const int tc = blockIdx.x * 64;
    const int tr = blockIdx.y * 64;
    const int t  = threadIdx.x;
    const int r  = t >> 2;
    const int c4 = (t & 3) * 16;

    const float* s = src + (size_t)(tr + r) * MOUT + tc + c4;
    #pragma unroll
    for (int j = 0; j < 16; j += 4) {
        float4 v = *(const float4*)(s + j);
        tile[r][c4 + j + 0] = v.x;
        tile[r][c4 + j + 1] = v.y;
        tile[r][c4 + j + 2] = v.z;
        tile[r][c4 + j + 3] = v.w;
    }
    __syncthreads();

    alignas(16) unsigned short tmp[16];
    #pragma unroll
    for (int j = 0; j < 16; ++j) tmp[j] = f2bf(tile[c4 + j][r]);
    unsigned short* d = dst + (size_t)(tc + r) * K + tr + c4;
    *(U4*)d       = *(const U4*)tmp;
    *(U4*)(d + 8) = *(const U4*)(tmp + 8);
}

// ---------------------------------------------------------------------------
// Kernel A3: wk/wv [64, 512] fp32 -> bf16 [h][oc(64)][k(64)] (transposed)
// ---------------------------------------------------------------------------
__global__ __launch_bounds__(256) void k_transpose_w(const float* __restrict__ src,
                                                     unsigned short* __restrict__ dst) {
    int i = blockIdx.x * blockDim.x + threadIdx.x;
    if (i >= HH * DD * DD) return;
    int h   = i >> 12;
    int rem = i & 4095;
    int oc  = rem >> 6;
    int k   = rem & 63;
    dst[i] = f2bf(src[k * (DD * HH) + h * DD + oc]);
}

// ---------------------------------------------------------------------------
// Kernel B: Fi = LayerNorm( gelu( X[1024,K] @ W[K,4096] ) ) as bf16.
// W pre-transposed as WT[4096][K]. Block tile 128x64, 4 waves; each wave owns
// 32 rows (2 M-tiles) x 64 cols, so every staged B tile feeds 8 WMMAs/wave
// and every B fragment is reused by 2 WMMAs. LN per 64-col row group via
// 16-lane butterfly. Double-buffered LDS + register prefetch.
// ---------------------------------------------------------------------------
#define LDG 40   // padded row stride (elems): 80B -> 20-bank stride, conflict-free
__global__ __launch_bounds__(128) void k_gemm_gelu_ln(
    const unsigned short* __restrict__ X,    // [1024, K] bf16
    const unsigned short* __restrict__ WT,   // [4096, K] bf16 (transposed)
    const float* __restrict__ gamma,         // [64]
    const float* __restrict__ beta,          // [64]
    unsigned short* __restrict__ fiOut,      // [1024, 8192] bf16
    int K, int colOffset)
{
    __shared__ alignas(32) unsigned short ldsA[2][128 * LDG];   // [row][k]
    __shared__ alignas(32) unsigned short ldsB[2][64 * LDG];    // [col][k]

    const int t    = threadIdx.x;
    const int lane = t & 31;
    const int w    = t >> 5;
    const int l15  = lane & 15;
    const int kh   = lane >> 4;
    const int mBase   = blockIdx.y * 128;
    const int colBase = blockIdx.x * 64;

    v8f acc[2][4];
    #pragma unroll
    for (int mi = 0; mi < 2; ++mi)
        #pragma unroll
        for (int nt = 0; nt < 4; ++nt) acc[mi][nt] = {};

    // staging: A -> thread t owns full row t (32 elems/step);
    //          B -> thread t owns (row t>>1, 16-elem chunk (t&1)*16)
    const unsigned short* pA = X + (size_t)(mBase + t) * K;
    const int brow = t >> 1;
    const int bko  = (t & 1) * 16;
    const unsigned short* pB = WT + (size_t)(colBase + brow) * K + bko;
    const int sIdxA = t * LDG;
    const int sIdxB = brow * LDG + bko;

    const int NK = K >> 5;

    // prologue: stage K-step 0 into buffer 0
    {
        U4 a0 = *(const U4*)pA,        a1 = *(const U4*)(pA + 8);
        U4 a2 = *(const U4*)(pA + 16), a3 = *(const U4*)(pA + 24);
        U4 b0 = *(const U4*)pB,        b1 = *(const U4*)(pB + 8);
        *(U4*)&ldsA[0][sIdxA]      = a0;  *(U4*)&ldsA[0][sIdxA + 8]  = a1;
        *(U4*)&ldsA[0][sIdxA + 16] = a2;  *(U4*)&ldsA[0][sIdxA + 24] = a3;
        *(U4*)&ldsB[0][sIdxB]      = b0;  *(U4*)&ldsB[0][sIdxB + 8]  = b1;
    }
    __syncthreads();

    for (int kt = 0; kt < NK; ++kt) {
        const bool more = (kt + 1 < NK);
        U4 na0, na1, na2, na3, nb0, nb1;
        if (more) {               // register prefetch of next K-step
            const unsigned short* qa = pA + (kt + 1) * 32;
            const unsigned short* qb = pB + (kt + 1) * 32;
            na0 = *(const U4*)qa;        na1 = *(const U4*)(qa + 8);
            na2 = *(const U4*)(qa + 16); na3 = *(const U4*)(qa + 24);
            nb0 = *(const U4*)qb;        nb1 = *(const U4*)(qb + 8);
            if (kt + 2 < NK) {    // cache prefetch two steps ahead
                __builtin_prefetch(pA + (kt + 2) * 32, 0, 1);
                __builtin_prefetch(pB + (kt + 2) * 32, 0, 1);
            }
        }

        // ---- compute on buffer kt&1: 2 A frags + 4 B frags -> 8 WMMAs ----
        const int cur = kt & 1;
        Frag a[2];
        #pragma unroll
        for (int mi = 0; mi < 2; ++mi) {
            const unsigned short* p = &ldsA[cur][(w * 32 + mi * 16 + l15) * LDG + kh * 8];
            a[mi].h[0] = *(const B16x8*)p;          // K = kh*8 .. +7
            a[mi].h[1] = *(const B16x8*)(p + 16);   // K = 16+kh*8 .. +7
        }
        #pragma unroll
        for (int nt = 0; nt < 4; ++nt) {
            Frag bfr;
            const unsigned short* p = &ldsB[cur][(nt * 16 + l15) * LDG + kh * 8];
            bfr.h[0] = *(const B16x8*)p;
            bfr.h[1] = *(const B16x8*)(p + 16);
            acc[0][nt] = __builtin_amdgcn_wmma_f32_16x16x32_bf16(
                false, a[0].v, false, bfr.v, (short)0, acc[0][nt], false, false);
            acc[1][nt] = __builtin_amdgcn_wmma_f32_16x16x32_bf16(
                false, a[1].v, false, bfr.v, (short)0, acc[1][nt], false, false);
        }

        if (more) {
            __syncthreads();   // everyone done reading buf[(kt+1)&1] from kt-1
            const int nxt = (kt + 1) & 1;
            *(U4*)&ldsA[nxt][sIdxA]      = na0;  *(U4*)&ldsA[nxt][sIdxA + 8]  = na1;
            *(U4*)&ldsA[nxt][sIdxA + 16] = na2;  *(U4*)&ldsA[nxt][sIdxA + 24] = na3;
            *(U4*)&ldsB[nxt][sIdxB]      = nb0;  *(U4*)&ldsB[nxt][sIdxB + 8]  = nb1;
            __syncthreads();
        }
    }

    // ---- exact GELU in place ----
    #pragma unroll
    for (int mi = 0; mi < 2; ++mi)
        #pragma unroll
        for (int nt = 0; nt < 4; ++nt)
            #pragma unroll
            for (int r = 0; r < 8; ++r)
                acc[mi][nt][r] = gelu_exact(acc[mi][nt][r]);

    // ---- fused LayerNorm over the 64-col group + bf16 store ----
    float gm[4], bt[4];
    #pragma unroll
    for (int nt = 0; nt < 4; ++nt) { gm[nt] = gamma[nt * 16 + l15]; bt[nt] = beta[nt * 16 + l15]; }

    #pragma unroll
    for (int mi = 0; mi < 2; ++mi) {
        #pragma unroll
        for (int r = 0; r < 8; ++r) {
            float s = 0.f, ss = 0.f;
            #pragma unroll
            for (int nt = 0; nt < 4; ++nt) { float g = acc[mi][nt][r]; s += g; ss += g * g; }
            #pragma unroll
            for (int m = 1; m < 16; m <<= 1) {
                s  += __shfl_xor(s,  m, 16);
                ss += __shfl_xor(ss, m, 16);
            }
            float mean = s * (1.0f / 64.0f);
            float var  = ss * (1.0f / 64.0f) - mean * mean;
            float rstd = rsqrtf(var + 1e-5f);
            int gRow = mBase + w * 32 + mi * 16 + r + (kh << 3);
            size_t rowOff = (size_t)gRow * 8192 + colOffset + colBase;
            #pragma unroll
            for (int nt = 0; nt < 4; ++nt) {
                float y = (acc[mi][nt][r] - mean) * rstd * gm[nt] + bt[nt];
                fiOut[rowOff + nt * 16 + l15] = f2bf(y);
            }
        }
    }
}

// ---------------------------------------------------------------------------
// Kernel C: per-(b,h) attention. Fi tile (16 KB) is DMA'd into LDS by the
// Tensor Data Mover with hardware row padding (pad 16B every 128B -> 144B
// rows); wk_h/wv_h staged manually. K,V computed via bf16 WMMA, K folded
// into QK and V into L1 norms in the epilogue, then softmax + S = A*V.
// ---------------------------------------------------------------------------
#define LDF 72   // padded row stride (elems): 144B -> 36-bank stride, conflict-free
__global__ __launch_bounds__(128) void k_attn(
    const unsigned short* __restrict__ Fi,   // [1024, 8192] bf16 ([b][n][d])
    const unsigned short* __restrict__ WkT,  // [8][64 oc][64 k] bf16
    const unsigned short* __restrict__ WvT,  // [8][64 oc][64 k] bf16
    const float* __restrict__ cmd,           // [1024, 128]
    const float* __restrict__ wq,            // [128, 512]
    float* __restrict__ out)                 // [1024, 8, 64]
{
    extern __shared__ char smem[];
    unsigned short* sFi = (unsigned short*)smem;  // [128 n][LDF]   18,432 B
    unsigned short* sWk = sFi + 128 * LDF;        // [64 oc][LDF]    9,216 B
    unsigned short* sWv = sWk + 64 * LDF;         //                 9,216 B
    unsigned short* sV  = sWv + 64 * LDF;         // [128 n][64 d]  16,384 B (bf16)
    float* sQ  = (float*)(sV + 8192);             // [64]
    float* sQK = sQ + 64;                         // [128]
    float* sMV = sQK + 128;                       // [128]
    float* sA  = sMV + 128;                       // [128]

    const int t    = threadIdx.x;
    const int lane = t & 31;
    const int w    = t >> 5;
    const int l15  = lane & 15;
    const int kh   = lane >> 4;
    const int h    = blockIdx.x;
    const int b    = blockIdx.y;

#if __has_builtin(__builtin_amdgcn_tensor_load_to_lds) && __has_builtin(__builtin_amdgcn_s_wait_tensorcnt)
    // --- TDM: DMA Fi[b] (8192 bf16, contiguous) into sFi with row padding ---
    // D# per CDNA5 ISA ch.8: 1-D tensor, data_size=2B, tile_dim0=8192,
    // pad_enable: pad_interval code 4 (every 32 DWORDs = 128B = one 64-elem
    // row), pad_amount code 3 (4 DWORDs = 16B) -> 144B LDS row stride == LDF.
    if (w == 0) {   // one TDM op per workgroup (wave 0 issues; EXEC ignored)
        unsigned ldsAddr = (unsigned)(size_t)(void*)sFi;   // low 32b = LDS offset
        unsigned long long ga = (unsigned long long)(size_t)(Fi + (size_t)b * 8192);
        u32x4 g0;
        g0[0] = 1u;                                        // count=1 (valid user D#)
        g0[1] = ldsAddr;                                   // lds_addr
        g0[2] = (unsigned)(ga & 0xffffffffu);              // global_addr[31:0]
        g0[3] = (unsigned)((ga >> 32) & 0x01ffffffu)       // global_addr[56:32]
              | (2u << 30);                                // type = 2 ("image")
        i32x8 g1;
        g1[0] = (int)((1u << 16)      // data_size = 2 bytes
                    | (1u << 20)      // pad_enable
                    | (4u << 22)      // pad_interval: 32 DWORDs before pad
                    | (3u << 25));    // pad_amount: 4 DWORDs of pad
        g1[1] = (int)(8192u << 16);   // tensor_dim0[15:0]=8192 (atomic_bar_addr=0)
        g1[2] = (int)(1u << 16);      // tensor_dim0[31:16]=0 | tensor_dim1=1
        g1[3] = (int)(8192u << 16);   // tensor_dim1 hi=0     | tile_dim0=8192
        g1[4] = 1;                    // tile_dim1=1, tile_dim2=0
        g1[5] = 8192;                 // tensor_dim0_stride (lo 32)
        g1[6] = 0;
        g1[7] = 0;
        i32x4 z4; z4[0] = 0; z4[1] = 0; z4[2] = 0; z4[3] = 0;   // <=2D: groups 2/3 unused
        i32x8 z8; z8[0] = 0; z8[1] = 0; z8[2] = 0; z8[3] = 0;
                  z8[4] = 0; z8[5] = 0; z8[6] = 0; z8[7] = 0;
        // this toolchain declares the 6-arg form:
        // (u32x4 g0, i32x8 g1, i32x4, i32x4, i32x8, i32 cpol)
        __builtin_amdgcn_tensor_load_to_lds(g0, g1, z4, z4, z8, 0);
        __builtin_amdgcn_s_wait_tensorcnt(0);
    }
#else
    // --- fallback: manual vectorized copy with padded rows ---
    {
        const unsigned short* src = Fi + (size_t)b * 8192 + t * 64;
        unsigned short* dst = sFi + t * LDF;
        #pragma unroll
        for (int j = 0; j < 8; ++j) *(U4*)(dst + j * 8) = *(const U4*)(src + j * 8);
    }
#endif

    // stage wk_h / wv_h: thread t owns (row = t>>1, 32-elem chunk = (t&1)*32)
    {
        const int r  = t >> 1;
        const int ko = (t & 1) * 32;
        const unsigned short* sk = WkT + h * 4096 + r * 64 + ko;
        const unsigned short* sv = WvT + h * 4096 + r * 64 + ko;
        unsigned short* dk = sWk + r * LDF + ko;
        unsigned short* dv = sWv + r * LDF + ko;
        #pragma unroll
        for (int j = 0; j < 4; ++j) {
            *(U4*)(dk + j * 8) = *(const U4*)(sk + j * 8);
            *(U4*)(dv + j * 8) = *(const U4*)(sv + j * 8);
        }
    }
    // Q[d] = command[b] . wq[:, h*64+d]   (128-MAC fp32 dot; tiny)
    if (t < 64) {
        const float* c = cmd + (size_t)b * N_CMD;
        const float* q = wq + h * DD + t;
        float acc = 0.f;
        #pragma unroll 4
        for (int i = 0; i < N_CMD; ++i) acc += c[i] * q[(size_t)i * (DD * HH)];
        sQ[t] = acc;
    }
    __syncthreads();

    float qv[4];
    #pragma unroll
    for (int nt = 0; nt < 4; ++nt) qv[nt] = sQ[nt * 16 + l15];

    // Each wave owns 2 M-tiles (32 of 128 feature rows), all 4 N-tiles (64 d).
    for (int mi = 0; mi < 2; ++mi) {
        const int mt = w * 2 + mi;
        v8f aK[4], aV[4];
        #pragma unroll
        for (int i = 0; i < 4; ++i) { aK[i] = {}; aV[i] = {}; }

        #pragma unroll
        for (int kk = 0; kk < 64; kk += 32) {
            Frag a;
            const unsigned short* pa = sFi + (mt * 16 + l15) * LDF + kk + kh * 8;
            a.h[0] = *(const B16x8*)pa;
            a.h[1] = *(const B16x8*)(pa + 16);
            Frag bk[4], bv[4];
            #pragma unroll
            for (int nt = 0; nt < 4; ++nt) {
                const unsigned short* pk = sWk + (nt * 16 + l15) * LDF + kk + kh * 8;
                const unsigned short* pv = sWv + (nt * 16 + l15) * LDF + kk + kh * 8;
                bk[nt].h[0] = *(const B16x8*)pk;  bk[nt].h[1] = *(const B16x8*)(pk + 16);
                bv[nt].h[0] = *(const B16x8*)pv;  bv[nt].h[1] = *(const B16x8*)(pv + 16);
            }
            #pragma unroll
            for (int nt = 0; nt < 4; ++nt)
                aK[nt] = __builtin_amdgcn_wmma_f32_16x16x32_bf16(
                    false, a.v, false, bk[nt].v, (short)0, aK[nt], false, false);
            #pragma unroll
            for (int nt = 0; nt < 4; ++nt)
                aV[nt] = __builtin_amdgcn_wmma_f32_16x16x32_bf16(
                    false, a.v, false, bv[nt].v, (short)0, aV[nt], false, false);
        }
        // Epilogue: QK[n] = sum_d Q[d]*K[n,d]; mV[n] = mean_d |V[n,d]|; stash V.
        #pragma unroll
        for (int r = 0; r < 8; ++r) {
            const int nrow = mt * 16 + r + (kh << 3);
            float pq = 0.f, pa = 0.f;
            #pragma unroll
            for (int nt = 0; nt < 4; ++nt) {
                float kvv = aK[nt][r];
                float vvv = aV[nt][r];
                pq += qv[nt] * kvv;
                pa += fabsf(vvv);
                sV[nrow * 64 + nt * 16 + l15] = f2bf(vvv);
            }
            #pragma unroll
            for (int m = 1; m < 16; m <<= 1) {
                pq += __shfl_xor(pq, m, 16);
                pa += __shfl_xor(pa, m, 16);
            }
            if (l15 == 0) { sQK[nrow] = pq; sMV[nrow] = pa * (1.0f / 64.0f); }
        }
    }
    __syncthreads();

    // softmax over n of (mV * QK / sqrt(D))
    sA[t] = sMV[t] * sQK[t] * 0.125f;
    __syncthreads();
    float mx = -3.402823466e38f;
    for (int i = 0; i < NFNF; ++i) mx = fmaxf(mx, sA[i]);
    float sum = 0.f;
    for (int i = 0; i < NFNF; ++i) sum += expf(sA[i] - mx);
    float aw = expf(sA[t] - mx) / sum;
    __syncthreads();
    sA[t] = aw;
    __syncthreads();

    // S[d] = sum_n A[n] * V[n,d]
    if (t < 64) {
        float s = 0.f;
        for (int n = 0; n < NFNF; ++n) s += sA[n] * bf2f(sV[n * 64 + t]);
        out[((size_t)b * HH + h) * DD + t] = s;
    }
}

// ---------------------------------------------------------------------------
// Workspace layout (bytes) — total ~48.4 MB of d_ws
// ---------------------------------------------------------------------------
static const size_t WZ_OFF  = 0;                                        // wzT [4096][2048]
static const size_t WH_OFF  = WZ_OFF  + (size_t)N_ENC * MOUT * 2;       // whT [4096][1024]
static const size_t FE_OFF  = WH_OFF  + (size_t)N_HID * MOUT * 2;
static const size_t HI_OFF  = FE_OFF  + (size_t)BB * N_ENC * 2;
static const size_t WKT_OFF = HI_OFF  + (size_t)BB * N_HID * 2;
static const size_t WVT_OFF = WKT_OFF + (size_t)HH * DD * DD * 2;
static const size_t FI_OFF  = WVT_OFF + (size_t)HH * DD * DD * 2;
// FI: 1024 * 8192 * 2 = 16 MB

extern "C" void kernel_launch(void* const* d_in, const int* in_sizes, int n_in,
                              void* d_out, int out_size, void* d_ws, size_t ws_size,
                              hipStream_t stream)
{
    const float* feature = (const float*)d_in[0];
    const float* hidden  = (const float*)d_in[1];
    const float* command = (const float*)d_in[2];
    const float* wz      = (const float*)d_in[3];
    const float* wh      = (const float*)d_in[4];
    const float* wq      = (const float*)d_in[5];
    const float* wk      = (const float*)d_in[6];
    const float* wv      = (const float*)d_in[7];
    const float* gamma   = (const float*)d_in[8];
    const float* beta    = (const float*)d_in[9];
    float* out = (float*)d_out;

    char* ws = (char*)d_ws;
    unsigned short* wzT = (unsigned short*)(ws + WZ_OFF);
    unsigned short* whT = (unsigned short*)(ws + WH_OFF);
    unsigned short* feB = (unsigned short*)(ws + FE_OFF);
    unsigned short* hiB = (unsigned short*)(ws + HI_OFF);
    unsigned short* wkT = (unsigned short*)(ws + WKT_OFF);
    unsigned short* wvT = (unsigned short*)(ws + WVT_OFF);
    unsigned short* fiB = (unsigned short*)(ws + FI_OFF);

    // 1) fp32 -> bf16 conversions; big weights are transposed to [col][k]
    k_cvt_T<<<dim3(MOUT / 64, N_ENC / 64), 256, 0, stream>>>(wz, wzT, N_ENC);
    k_cvt_T<<<dim3(MOUT / 64, N_HID / 64), 256, 0, stream>>>(wh, whT, N_HID);
    {
        int n;
        n = BB * N_ENC;  k_cvt<<<(n + 255) / 256, 256, 0, stream>>>(feature, feB, n);
        n = BB * N_HID;  k_cvt<<<(n + 255) / 256, 256, 0, stream>>>(hidden, hiB, n);
    }
    k_transpose_w<<<(HH * DD * DD + 255) / 256, 256, 0, stream>>>(wk, wkT);
    k_transpose_w<<<(HH * DD * DD + 255) / 256, 256, 0, stream>>>(wv, wvT);

    // 2) Fi = LN(gelu(X @ W)) — two GEMMs writing halves of Fi[B,128,64]
    dim3 gB(MOUT / 64, BB / 128);   // (64, 8), 128x64 block tiles
    k_gemm_gelu_ln<<<gB, 128, 0, stream>>>(feB, wzT, gamma, beta, fiB, N_ENC, 0);
    k_gemm_gelu_ln<<<gB, 128, 0, stream>>>(hiB, whT, gamma, beta, fiB, N_HID, MOUT);

    // 3) attention: one block per (b,h); 55,040 B dynamic LDS (< 64 KB default)
    size_t smemC = (size_t)(128 * LDF + 64 * LDF + 64 * LDF + 8192) * 2
                 + (size_t)(64 + 128 + 128 + 128) * 4;
    k_attn<<<dim3(HH, BB), 128, smemC, stream>>>(fiB, wkT, wvT, command, wq, out);
}
